// CausalMultiHeadedAttention_24653112279165
// MI455X (gfx1250) — matrix operand order repaired and checked by reading, not compiled, not run on hardware
//
#include <hip/hip_runtime.h>
#include <hip/hip_bf16.h>

// ---------------------------------------------------------------------------
// CDNA5 (gfx1250) causal multi-headed attention, f16 WMMA everywhere.
// wave32, one wave per block for all WMMA kernels.
// GEMM waves compute 32x64 output tiles (2 M-subtiles x 4 N-subtiles) so each
// B tile feeds two v_wmma ops; all operand loads are batched ahead of the
// WMMA group so the compiler can clause them and amortize s_wait_loadcnt.
// ---------------------------------------------------------------------------

#define SEQ   2048
#define BATCH 4
#define DMODEL 512
#define NHEADS 8
#define DK    64

typedef __attribute__((ext_vector_type(16))) _Float16 h16;
typedef __attribute__((ext_vector_type(8)))  _Float16 h8;
typedef __attribute__((ext_vector_type(8)))  float    f8v;

__device__ __forceinline__ f8v zero8() {
  f8v z;
#pragma unroll
  for (int i = 0; i < 8; ++i) z[i] = 0.0f;
  return z;
}

__device__ __forceinline__ f8v wmma16(h16 a, h16 b, f8v c) {
  return __builtin_amdgcn_wmma_f32_16x16x32_f16(false, a, false, b,
                                                (short)0, c, false, false);
}

// A operand (16x32 f16, M x K): lane L holds row M=L&15; halves i<8 -> K=g*8+i,
// halves i>=8 -> K=16+g*8+(i-8), g = L>>4.  Source row-major, leading dim lda.
__device__ __forceinline__ h16 load_a16(const _Float16* __restrict__ base,
                                        int lda, int m0, int k0, int lane) {
  int M = lane & 15, g = lane >> 4;
  const _Float16* p = base + (size_t)(m0 + M) * lda + k0 + g * 8;
  h8 lo = *(const h8*)p;
  h8 hi = *(const h8*)(p + 16);
  h16 a;
#pragma unroll
  for (int i = 0; i < 8; ++i) { a[i] = lo[i]; a[8 + i] = hi[i]; }
  return a;
}

// B operand (32x16 f16, K x N) loaded from B^T (N-major, leading dim ldb over K):
// lane L holds column N=L&15; halves i<8 -> K=g*8+i, halves i>=8 -> K=16+g*8+(i-8).
__device__ __forceinline__ h16 load_b16(const _Float16* __restrict__ bt,
                                        int ldb, int n0, int k0, int lane) {
  int N = lane & 15, g = lane >> 4;
  const _Float16* p = bt + (size_t)(n0 + N) * ldb + k0 + g * 8;
  h8 lo = *(const h8*)p;
  h8 hi = *(const h8*)(p + 16);
  h16 b;
#pragma unroll
  for (int i = 0; i < 8; ++i) { b[i] = lo[i]; b[8 + i] = hi[i]; }
  return b;
}

// ---------------------------------------------------------------------------
// Prep kernels
// ---------------------------------------------------------------------------
__global__ void cvt_f32_to_f16_kernel(const float* __restrict__ src,
                                      _Float16* __restrict__ dst, int n) {
  for (int i = blockIdx.x * blockDim.x + threadIdx.x; i < n;
       i += gridDim.x * blockDim.x)
    dst[i] = (_Float16)src[i];
}

// conv weight [co][ci][t] (t fastest) -> f16 [co][t*512+ci]
__global__ void convw_transpose_kernel(const float* __restrict__ src,
                                       _Float16* __restrict__ dst) {
  int n = DMODEL * DMODEL * 3;
  for (int i = blockIdx.x * blockDim.x + threadIdx.x; i < n;
       i += gridDim.x * blockDim.x) {
    int co = i / (DMODEL * 3);
    int rem = i - co * (DMODEL * 3);
    int t  = rem / DMODEL;
    int ci = rem - t * DMODEL;
    dst[i] = (_Float16)src[((size_t)co * DMODEL + ci) * 3 + t];
  }
}

// ---------------------------------------------------------------------------
// Causal conv1d (K=3) as GEMM: K-dim = 3*512 = 1536.
// x: f16 [S][B][512]; wT: f16 [512][1536]; out: f16 [B][8][S][64]
// grid: (S/32, B, D/64), block 32 (one wave), tile 32x64.
// ---------------------------------------------------------------------------
__global__ void __launch_bounds__(32)
conv_qk_kernel(const _Float16* __restrict__ x, const _Float16* __restrict__ wT,
               const float* __restrict__ bias, _Float16* __restrict__ out) {
  int lane = threadIdx.x;
  int s0 = blockIdx.x * 32;
  int b  = blockIdx.y;
  int z  = blockIdx.z;           // 64-channel tile == head index
  int M = lane & 15, g = lane >> 4;

  f8v c[2][4];
#pragma unroll
  for (int mt = 0; mt < 2; ++mt)
#pragma unroll
    for (int nt = 0; nt < 4; ++nt) c[mt][nt] = zero8();

  for (int kc = 0; kc < 48; ++kc) {
    int tap = kc >> 4;           // 0..2, 16 chunks per tap
    int ci0 = (kc & 15) * 32;
    int k0  = kc * 32;
    // A tile 0 (rows s0..s0+15): may need causal zero-pad at s0 == 0
    int srow0 = s0 + M + tap - 2;
    h16 a0;
    if (srow0 >= 0) {
      const _Float16* p = x + ((size_t)srow0 * BATCH + b) * DMODEL + ci0 + g * 8;
      h8 lo = *(const h8*)p;
      h8 hi = *(const h8*)(p + 16);
#pragma unroll
      for (int i = 0; i < 8; ++i) { a0[i] = lo[i]; a0[8 + i] = hi[i]; }
    } else {
#pragma unroll
      for (int i = 0; i < 16; ++i) a0[i] = (_Float16)0.0f;
    }
    // A tile 1 (rows s0+16..s0+31): srow >= 14 always, no pad
    h16 a1;
    {
      const _Float16* p =
          x + ((size_t)(srow0 + 16) * BATCH + b) * DMODEL + ci0 + g * 8;
      h8 lo = *(const h8*)p;
      h8 hi = *(const h8*)(p + 16);
#pragma unroll
      for (int i = 0; i < 8; ++i) { a1[i] = lo[i]; a1[8 + i] = hi[i]; }
    }
    // batch all four B tiles, then 8 WMMAs
    h16 bb[4];
#pragma unroll
    for (int nt = 0; nt < 4; ++nt)
      bb[nt] = load_b16(wT, 1536, z * 64 + nt * 16, k0, lane);
#pragma unroll
    for (int nt = 0; nt < 4; ++nt) {
      c[0][nt] = wmma16(a0, bb[nt], c[0][nt]);
      c[1][nt] = wmma16(a1, bb[nt], c[1][nt]);
    }
  }
  // store head-major [B][h][S][64]
#pragma unroll
  for (int mt = 0; mt < 2; ++mt)
#pragma unroll
    for (int nt = 0; nt < 4; ++nt)
#pragma unroll
      for (int r = 0; r < 8; ++r) {
        int co = z * 64 + nt * 16 + M;
        int s  = s0 + mt * 16 + r + 8 * g;
        float v = c[mt][nt][r] + bias[co];
        out[(((size_t)b * NHEADS + z) * SEQ + s) * DK + (co & 63)] = (_Float16)v;
      }
}

// ---------------------------------------------------------------------------
// V projection: v @ lin1_w.T + b, stored TRANSPOSED: VT f16 [B][8][64][S]
// x: f16 [S][B][512]; w: f16 [512][512] (B^T layout). grid (S/32, B, 8).
// ---------------------------------------------------------------------------
__global__ void __launch_bounds__(32)
vproj_kernel(const _Float16* __restrict__ x, const _Float16* __restrict__ w,
             const float* __restrict__ bias, _Float16* __restrict__ vt) {
  int lane = threadIdx.x;
  int s0 = blockIdx.x * 32;
  int b  = blockIdx.y;
  int z  = blockIdx.z;
  int M = lane & 15, g = lane >> 4;

  f8v c[2][4];
#pragma unroll
  for (int mt = 0; mt < 2; ++mt)
#pragma unroll
    for (int nt = 0; nt < 4; ++nt) c[mt][nt] = zero8();

  for (int kc = 0; kc < 16; ++kc) {
    int k0 = kc * 32;
    h16 a0, a1;
    {
      const _Float16* p =
          x + ((size_t)(s0 + M) * BATCH + b) * DMODEL + k0 + g * 8;
      h8 lo = *(const h8*)p;
      h8 hi = *(const h8*)(p + 16);
#pragma unroll
      for (int i = 0; i < 8; ++i) { a0[i] = lo[i]; a0[8 + i] = hi[i]; }
    }
    {
      const _Float16* p =
          x + ((size_t)(s0 + 16 + M) * BATCH + b) * DMODEL + k0 + g * 8;
      h8 lo = *(const h8*)p;
      h8 hi = *(const h8*)(p + 16);
#pragma unroll
      for (int i = 0; i < 8; ++i) { a1[i] = lo[i]; a1[8 + i] = hi[i]; }
    }
    h16 bb[4];
#pragma unroll
    for (int nt = 0; nt < 4; ++nt)
      bb[nt] = load_b16(w, DMODEL, z * 64 + nt * 16, k0, lane);
#pragma unroll
    for (int nt = 0; nt < 4; ++nt) {
      c[0][nt] = wmma16(a0, bb[nt], c[0][nt]);
      c[1][nt] = wmma16(a1, bb[nt], c[1][nt]);
    }
  }
#pragma unroll
  for (int mt = 0; mt < 2; ++mt)
#pragma unroll
    for (int nt = 0; nt < 4; ++nt)
#pragma unroll
      for (int r = 0; r < 8; ++r) {
        int co = z * 64 + nt * 16 + M;
        int s  = s0 + mt * 16 + r + 8 * g;
        float v = c[mt][nt][r] + bias[co];
        vt[(((size_t)b * NHEADS + z) * DK + (co & 63)) * SEQ + s] = (_Float16)v;
      }
}

// ---------------------------------------------------------------------------
// Flash attention: per (q-tile of 16, b, h).  Online softmax over 32-wide key
// blocks.  Q,K: f16 [B][8][S][64]; VT: f16 [B][8][64][S]; X out: f16 [B*S][512]
// ---------------------------------------------------------------------------
__global__ void __launch_bounds__(32)
attn_kernel(const _Float16* __restrict__ Qp, const _Float16* __restrict__ Kp,
            const _Float16* __restrict__ VT, _Float16* __restrict__ X) {
  __shared__ _Float16 pbuf[16 * 40];   // 16x32 P tile, row stride 40 halves
  int lane = threadIdx.x;
  int s0 = blockIdx.x * 16;
  int b  = blockIdx.y;
  int hd = blockIdx.z;
  int M = lane & 15, g = lane >> 4;

  const _Float16* Qb = Qp + ((size_t)b * NHEADS + hd) * SEQ * DK;
  const _Float16* Kb = Kp + ((size_t)b * NHEADS + hd) * SEQ * DK;
  const _Float16* Vb = VT + ((size_t)b * NHEADS + hd) * DK * SEQ;

  h16 aq0 = load_a16(Qb, DK, s0, 0, lane);
  h16 aq1 = load_a16(Qb, DK, s0, 32, lane);

  f8v acc[4];
#pragma unroll
  for (int t = 0; t < 4; ++t) acc[t] = zero8();
  float mrow[8], lrow[8];
#pragma unroll
  for (int r = 0; r < 8; ++r) { mrow[r] = -1e30f; lrow[r] = 0.0f; }

  int nblocks = (s0 + 15) / 32 + 1;
  for (int kb = 0; kb < nblocks; ++kb) {
    int kbase = kb * 32;
    // ---- S = Q K^T: batch the 4 K tiles, then 4 WMMAs ----
    h16 bk[4];
    bk[0] = load_b16(Kb, DK, kbase,      0,  lane);
    bk[1] = load_b16(Kb, DK, kbase,      32, lane);
    bk[2] = load_b16(Kb, DK, kbase + 16, 0,  lane);
    bk[3] = load_b16(Kb, DK, kbase + 16, 32, lane);
    f8v sc0 = zero8(), sc1 = zero8();
    sc0 = wmma16(aq0, bk[0], sc0);
    sc0 = wmma16(aq1, bk[1], sc0);
    sc1 = wmma16(aq0, bk[2], sc1);
    sc1 = wmma16(aq1, bk[3], sc1);
    // ---- prefetch V tiles for this block (overlaps softmax math) ----
    h16 bv[4];
#pragma unroll
    for (int t = 0; t < 4; ++t)
      bv[t] = load_b16(Vb, SEQ, t * 16, kbase, lane);
    // ---- scale + causal mask + online softmax ----
    int qrow0 = s0 + 8 * g;
    int col0 = kbase + M, col1 = kbase + 16 + M;
#pragma unroll
    for (int r = 0; r < 8; ++r) {
      int qi = qrow0 + r;
      float v0 = sc0[r] * 0.125f; if (col0 > qi) v0 = -1e30f;
      float v1 = sc1[r] * 0.125f; if (col1 > qi) v1 = -1e30f;
      float bm = fmaxf(v0, v1);
      bm = fmaxf(bm, __shfl_xor(bm, 1, 32));
      bm = fmaxf(bm, __shfl_xor(bm, 2, 32));
      bm = fmaxf(bm, __shfl_xor(bm, 4, 32));
      bm = fmaxf(bm, __shfl_xor(bm, 8, 32));
      float mnew  = fmaxf(mrow[r], bm);
      float alpha = __expf(mrow[r] - mnew);
      float p0 = __expf(v0 - mnew);
      float p1 = __expf(v1 - mnew);
      float rs = p0 + p1;
      rs += __shfl_xor(rs, 1, 32);
      rs += __shfl_xor(rs, 2, 32);
      rs += __shfl_xor(rs, 4, 32);
      rs += __shfl_xor(rs, 8, 32);
      lrow[r] = lrow[r] * alpha + rs;
      mrow[r] = mnew;
#pragma unroll
      for (int t = 0; t < 4; ++t) acc[t][r] *= alpha;
      pbuf[(r + 8 * g) * 40 + M]      = (_Float16)p0;
      pbuf[(r + 8 * g) * 40 + 16 + M] = (_Float16)p1;
    }
    __syncthreads();
    // ---- reload P in A layout (LDS transpose) ----
    h16 ap;
    {
      const _Float16* p = pbuf + M * 40 + g * 8;
      h8 lo = *(const h8*)p;
      h8 hi = *(const h8*)(p + 16);
#pragma unroll
      for (int i = 0; i < 8; ++i) { ap[i] = lo[i]; ap[8 + i] = hi[i]; }
    }
    // ---- O += P V  (K-dim = 32 keys, N = dk in 4 tiles) ----
#pragma unroll
    for (int t = 0; t < 4; ++t) acc[t] = wmma16(ap, bv[t], acc[t]);
    __syncthreads();
  }
  // ---- epilogue: normalize and scatter into X [B*S][512] ----
  float inv[8];
#pragma unroll
  for (int r = 0; r < 8; ++r) inv[r] = 1.0f / lrow[r];
#pragma unroll
  for (int t = 0; t < 4; ++t)
#pragma unroll
    for (int r = 0; r < 8; ++r) {
      int s = s0 + r + 8 * g;
      size_t row = (size_t)b * SEQ + s;
      X[row * DMODEL + hd * DK + t * 16 + M] = (_Float16)(acc[t][r] * inv[r]);
    }
}

// ---------------------------------------------------------------------------
// Output projection: out = X @ lin2_w.T + b, fp32, stored [S][B][512].
// grid (B*S/32, D/64), tile 32x64.
// ---------------------------------------------------------------------------
__global__ void __launch_bounds__(32)
outproj_kernel(const _Float16* __restrict__ X, const _Float16* __restrict__ w,
               const float* __restrict__ bias, float* __restrict__ out) {
  int lane = threadIdx.x;
  int t0 = blockIdx.x * 32;
  int z  = blockIdx.y;
  int M = lane & 15, g = lane >> 4;

  f8v c[2][4];
#pragma unroll
  for (int mt = 0; mt < 2; ++mt)
#pragma unroll
    for (int nt = 0; nt < 4; ++nt) c[mt][nt] = zero8();

  for (int kc = 0; kc < 16; ++kc) {
    int k0 = kc * 32;
    h16 a0 = load_a16(X, DMODEL, t0,      k0, lane);
    h16 a1 = load_a16(X, DMODEL, t0 + 16, k0, lane);
    h16 bb[4];
#pragma unroll
    for (int nt = 0; nt < 4; ++nt)
      bb[nt] = load_b16(w, DMODEL, z * 64 + nt * 16, k0, lane);
#pragma unroll
    for (int nt = 0; nt < 4; ++nt) {
      c[0][nt] = wmma16(a0, bb[nt], c[0][nt]);
      c[1][nt] = wmma16(a1, bb[nt], c[1][nt]);
    }
  }
#pragma unroll
  for (int mt = 0; mt < 2; ++mt)
#pragma unroll
    for (int nt = 0; nt < 4; ++nt)
#pragma unroll
      for (int r = 0; r < 8; ++r) {
        int token = t0 + mt * 16 + r + 8 * g;
        int bb2 = token >> 11;       // token / SEQ
        int ss  = token & (SEQ - 1);
        int co  = z * 64 + nt * 16 + M;
        out[((size_t)ss * BATCH + bb2) * DMODEL + co] = c[mt][nt][r] + bias[co];
      }
}

// ---------------------------------------------------------------------------
// Host side
// ---------------------------------------------------------------------------
static inline size_t align256(size_t x) { return (x + 255) & ~(size_t)255; }

extern "C" void kernel_launch(void* const* d_in, const int* in_sizes, int n_in,
                              void* d_out, int out_size, void* d_ws, size_t ws_size,
                              hipStream_t stream) {
  const float* q_in    = (const float*)d_in[0];
  const float* k_in    = (const float*)d_in[1];
  const float* v_in    = (const float*)d_in[2];
  // d_in[3] = attn_mask (causal; computed analytically, unused)
  const float* conv1_w = (const float*)d_in[4];
  const float* conv1_b = (const float*)d_in[5];
  const float* conv2_w = (const float*)d_in[6];
  const float* conv2_b = (const float*)d_in[7];
  const float* lin1_w  = (const float*)d_in[8];
  const float* lin1_b  = (const float*)d_in[9];
  const float* lin2_w  = (const float*)d_in[10];
  const float* lin2_b  = (const float*)d_in[11];
  float* out = (float*)d_out;

  const size_t NTOK = (size_t)SEQ * BATCH;           // 8192
  const size_t NQKV = NTOK * DMODEL;                 // 4 Mi elements
  char* ws = (char*)d_ws;
  size_t off = 0;
  _Float16* qf16 = (_Float16*)(ws + off); off = align256(off + NQKV * 2);
  _Float16* kf16 = (_Float16*)(ws + off); off = align256(off + NQKV * 2);
  _Float16* vf16 = (_Float16*)(ws + off); off = align256(off + NQKV * 2);
  _Float16* w1t  = (_Float16*)(ws + off); off = align256(off + (size_t)DMODEL * 1536 * 2);
  _Float16* w2t  = (_Float16*)(ws + off); off = align256(off + (size_t)DMODEL * 1536 * 2);
  _Float16* l1h  = (_Float16*)(ws + off); off = align256(off + (size_t)DMODEL * DMODEL * 2);
  _Float16* l2h  = (_Float16*)(ws + off); off = align256(off + (size_t)DMODEL * DMODEL * 2);
  _Float16* Qp   = (_Float16*)(ws + off); off = align256(off + NQKV * 2);
  _Float16* Kp   = (_Float16*)(ws + off); off = align256(off + NQKV * 2);
  _Float16* Vt   = (_Float16*)(ws + off); off = align256(off + NQKV * 2);
  _Float16* Xb   = (_Float16*)(ws + off); off = align256(off + NQKV * 2);
  (void)ws_size; (void)n_in; (void)in_sizes; (void)out_size;

  // 1) precision conversion / weight repack
  cvt_f32_to_f16_kernel<<<2048, 256, 0, stream>>>(q_in, qf16, (int)NQKV);
  cvt_f32_to_f16_kernel<<<2048, 256, 0, stream>>>(k_in, kf16, (int)NQKV);
  cvt_f32_to_f16_kernel<<<2048, 256, 0, stream>>>(v_in, vf16, (int)NQKV);
  convw_transpose_kernel<<<1024, 256, 0, stream>>>(conv1_w, w1t);
  convw_transpose_kernel<<<1024, 256, 0, stream>>>(conv2_w, w2t);
  cvt_f32_to_f16_kernel<<<512, 256, 0, stream>>>(lin1_w, l1h, DMODEL * DMODEL);
  cvt_f32_to_f16_kernel<<<512, 256, 0, stream>>>(lin2_w, l2h, DMODEL * DMODEL);

  // 2) projections (WMMA GEMMs, 32x64 tiles)
  dim3 gp(SEQ / 32, BATCH, DMODEL / 64);
  conv_qk_kernel<<<gp, 32, 0, stream>>>(qf16, w1t, conv1_b, Qp);
  conv_qk_kernel<<<gp, 32, 0, stream>>>(kf16, w2t, conv2_b, Kp);
  vproj_kernel <<<gp, 32, 0, stream>>>(vf16, l1h, lin1_b, Vt);

  // 3) flash attention
  dim3 ga(SEQ / 16, BATCH, NHEADS);
  attn_kernel<<<ga, 32, 0, stream>>>(Qp, Kp, Vt, Xb);

  // 4) output projection
  dim3 go((unsigned)(NTOK / 32), DMODEL / 64);
  outproj_kernel<<<go, 32, 0, stream>>>(Xb, l2h, lin2_b, out);
}
